// EnhancedUberCRSN_24902220382685
// MI455X (gfx1250) — matrix operand order repaired and checked by reading, not compiled, not run on hardware
//
#include <hip/hip_runtime.h>

// ---------------------------------------------------------------------------
// EnhancedUberCRSN for MI455X (gfx1250, wave32, WMMA + TDM async tensor loads)
// B=4096, D=256, S=64, N=4096
// ---------------------------------------------------------------------------

#define B_ 4096
#define D_ 256
#define S_ 64
#define N_ 4096
#define TWO_D 512

typedef __attribute__((ext_vector_type(16))) __bf16 bf16x16;
typedef __attribute__((ext_vector_type(8)))  float  v8f;
typedef __attribute__((ext_vector_type(4)))  unsigned int u32x4;
typedef __attribute__((ext_vector_type(4)))  int i32x4;
typedef __attribute__((ext_vector_type(8)))  int i32x8;

union FragBF {
    bf16x16 v;
    u32x4   q[2];
};

__device__ __forceinline__ unsigned short f2bf(float x) {
    unsigned int u = __float_as_uint(x);
    unsigned int r = u + 0x7FFFu + ((u >> 16) & 1u);   // round-to-nearest-even
    return (unsigned short)(r >> 16);
}
__device__ __forceinline__ float bf2f(unsigned short h) {
    return __uint_as_float(((unsigned int)h) << 16);
}

// A-fragment, 16-bit elements, 16x32 tile (ISA 7.12.2):
// lane<16: row=lane,   K-chunks {k0..k0+7},{k0+16..k0+23}
// lane>=16: row=lane-16,K-chunks {k0+8..k0+15},{k0+24..k0+31}
__device__ __forceinline__ FragBF load_a_frag(const unsigned short* base, int ld, int lane) {
    FragBF f;
    const unsigned short* p = base + (lane & 15) * ld + ((lane >> 4) << 3);
    f.q[0] = *(const u32x4*)(p);
    f.q[1] = *(const u32x4*)(p + 16);
    return f;
}
// B-fragment, 16-bit, 32x16 tile: lane holds column n=lane&15, same K-chunks as A.
__device__ __forceinline__ FragBF load_b_frag(const unsigned short* base, int ld, int lane) {
    FragBF f;
    const unsigned short* p = base + (lane & 15) * ld + ((lane >> 4) << 3);
    f.q[0] = *(const u32x4*)(p);
    f.q[1] = *(const u32x4*)(p + 16);
    return f;
}
__device__ __forceinline__ v8f wmma_bf16(const FragBF& a, const FragBF& b, v8f c) {
    return __builtin_amdgcn_wmma_f32_16x16x32_bf16(false, a.v, false, b.v,
                                                   (short)0, c, false, false);
}

// ---------------------------------------------------------------------------
// Tensor Data Mover: async 2D tile load (64 rows x 32 bf16 cols) into LDS,
// with LDS padding of 4 DWORDs every 16 DWORDs (row stride 40 halves = 80 B)
// to spread B-fragment reads across banks. Descriptor per ISA 8.3/8.4.
// ---------------------------------------------------------------------------
#define TDM_TILE_K 32                    // tile_dim0 (elements)
#define TDM_TILE_R 64                    // tile_dim1 (rows)
#define LDT 40                           // padded LDS row stride in halves

__device__ __forceinline__ unsigned int lds_offset(const void* p) {
    return (unsigned int)(unsigned long long)(p);   // low 32 bits = LDS byte offset
}

__device__ __forceinline__ void tdm_load_2d_bf16(unsigned int lds_off,
                                                 const unsigned short* gptr,
                                                 unsigned int stride_elems) {
    unsigned long long ga = (unsigned long long)(void*)gptr;
    u32x4 g0;
    g0[0] = 1u;                                           // count=1, user mode
    g0[1] = lds_off;                                      // lds_addr (bytes)
    g0[2] = (unsigned int)ga;                             // global_addr[31:0]
    g0[3] = (unsigned int)((ga >> 32) & 0x1FFFFFFull)     // global_addr[56:32]
            | (2u << 30);                                 // type = 2 ("image")
    i32x8 g1;
    // data_size=1 (2B), pad_enable=1, pad_interval=3 (16 DW), pad_amount=3 (4 DW)
    g1[0] = (int)0x06D10000;
    g1[1] = (int)((N_ & 0xFFFF) << 16);                   // tensor_dim0 lo16 << 16
    g1[2] = (int)((N_ >> 16) | ((N_ & 0xFFFF) << 16));    // dim0 hi16 | dim1 lo16
    g1[3] = (int)((N_ >> 16) | (TDM_TILE_K << 16));       // dim1 hi16 | tile_dim0
    g1[4] = (int)TDM_TILE_R;                              // tile_dim1 | tile_dim2=0
    g1[5] = (int)stride_elems;                            // tensor_dim0_stride lo32
    g1[6] = 0;                                            // stride0 hi16 | stride1 lo16
    g1[7] = 0;                                            // stride1 hi32
    i32x4 z4 = {0, 0, 0, 0};
#if __clang_major__ >= 23
    i32x8 z8 = {0, 0, 0, 0, 0, 0, 0, 0};
    __builtin_amdgcn_tensor_load_to_lds(g0, g1, z4, z4, z8, 0);
#else
    __builtin_amdgcn_tensor_load_to_lds(g0, g1, z4, z4, 0);
#endif
}

// ---------------------------------------------------------------------------
// Workspace layout (byte offsets; every region size is a multiple of 256B)
// ---------------------------------------------------------------------------
#define OFF_WBF     ((size_t)0)                       // 2 proj x 3 kinds x 256x256 bf16
#define OFF_ADJT    (OFF_WBF   + 786432)              // adjacency^T bf16  N*N*2
#define OFF_PD      (OFF_ADJT  + 33554432)            // prev_dist  bf16   B*N*2
#define OFF_PROBS   (OFF_PD    + 33554432)            // probs      bf16   B*N*2
#define OFF_CB      (OFF_PROBS + 33554432)            // codebook   bf16   N*512*2
#define OFF_CBT     (OFF_CB    + 4194304)             // codebook^T bf16   512*N*2
#define OFF_RBF     (OFF_CBT   + 4194304)             // read bf16         B*512*2
#define OFF_RF      (OFF_RBF   + 4194304)             // read f32          B*512*4
#define OFF_CNORM   (OFF_RF    + 8388608)             // ||codebook_n||^2  N*4
#define OFF_RNORM   (OFF_CNORM + 16384)               // ||read_b||^2      B*4
#define OFF_SCAL    (OFF_RNORM + 16384)               // [magsum, magsumsq, up]
#define OFF_LOGITS  (OFF_SCAL  + 256)                 // -d_total f32      B*N*4

// ---------------------------------------------------------------------------
// Prep kernels
// ---------------------------------------------------------------------------
__global__ __launch_bounds__(1) void k_zero_scalars(float* scal) {
    scal[0] = 0.0f; scal[1] = 0.0f; scal[2] = 0.0f;
}

// wbf layout: [proj(q=0,k=1)][kind(0=Wr,1=Wi,2=-Wi)][n*256+k], bf16
__global__ __launch_bounds__(256) void k_prep_weights(
        const float* Wq_r, const float* Wq_i,
        const float* Wk_r, const float* Wk_i, unsigned short* wbf) {
    int idx = blockIdx.x * 256 + threadIdx.x;
    if (idx >= 2 * 3 * 65536) return;
    int proj = idx / 196608;
    int rem  = idx % 196608;
    int kind = rem / 65536;
    int e    = rem % 65536;
    const float* R = proj ? Wk_r : Wq_r;
    const float* I = proj ? Wk_i : Wq_i;
    float v = (kind == 0) ? R[e] : ((kind == 1) ? I[e] : -I[e]);
    wbf[idx] = f2bf(v);
}

__global__ __launch_bounds__(256) void k_prep_pd(const float* pd, unsigned short* pd_bf) {
    size_t idx = (size_t)blockIdx.x * 256 + threadIdx.x;
    pd_bf[idx] = f2bf(pd[idx]);
}

// tiled transpose: adjT[n][k] = adjacency[k][n] (bf16)
__global__ __launch_bounds__(256) void k_prep_adjT(const float* adj, unsigned short* adjT) {
    __shared__ float tile[64][65];
    int n0 = blockIdx.x * 64, k0 = blockIdx.y * 64;
    int tid = threadIdx.x;
    #pragma unroll
    for (int i = 0; i < 16; ++i) {
        int lin = i * 256 + tid;
        int r = lin >> 6, c = lin & 63;
        tile[r][c] = adj[(size_t)(k0 + r) * N_ + (n0 + c)];
    }
    __syncthreads();
    #pragma unroll
    for (int i = 0; i < 16; ++i) {
        int lin = i * 256 + tid;
        int r = lin >> 6, c = lin & 63;
        adjT[(size_t)(n0 + r) * N_ + (k0 + c)] = f2bf(tile[c][r]);
    }
}

__global__ __launch_bounds__(256) void k_prep_cb(const float* cb,
        unsigned short* cb_bf, unsigned short* cbT_bf, float* cnorm) {
    __shared__ float red[256];
    int n = blockIdx.x, tid = threadIdx.x;
    float ss = 0.0f;
    for (int c = tid; c < TWO_D; c += 256) {
        float v = cb[(size_t)n * TWO_D + c];
        unsigned short h = f2bf(v);
        cb_bf[(size_t)n * TWO_D + c] = h;
        cbT_bf[(size_t)c * N_ + n]   = h;
        ss += v * v;
    }
    red[tid] = ss; __syncthreads();
    for (int st = 128; st > 0; st >>= 1) {
        if (tid < st) red[tid] += red[tid + st];
        __syncthreads();
    }
    if (tid == 0) cnorm[n] = red[0];
}

// ---------------------------------------------------------------------------
// Fused per-batch: mem update -> K proj -> Q tiles -> scores -> softmax ->
// w = attn^T ptr -> u = w@m -> read = clin(u, Wv) -> mag reductions.
// One workgroup (8 waves) per batch; ~151 KB LDS (< 160 KB CU-mode budget).
// ---------------------------------------------------------------------------
__global__ __launch_bounds__(256) void k_batch_attn(
        const float* z_real, const float* z_imag, const float* mem,
        const float* ptrv, const float* ctrl,
        const unsigned short* wbf, const float* Wv_r, const float* Wv_i,
        float* readf, unsigned short* readbf, float* rnorm2, float* magacc) {
    __shared__ unsigned short sM[S_ * TWO_D];   // updated mem  bf16 (64 KB)
    __shared__ unsigned short sK[S_ * TWO_D];   // Kf           bf16 (64 KB)
    __shared__ unsigned short sQ[16 * TWO_D];   // one Q tile   bf16 (16 KB)
    __shared__ float sS[16 * 64];               // score/attn tile (4 KB)
    __shared__ float sW[64];
    __shared__ float sPtr[64];
    __shared__ float sU[TWO_D];
    __shared__ float sRed[256];

    const int b = blockIdx.x;
    const int tid = threadIdx.x;
    const int lane = tid & 31;
    const int wave = tid >> 5;

    float g0 = 1.0f / (1.0f + __expf(-ctrl[b * 3 + 0]));
    float g1 = 1.0f / (1.0f + __expf(-ctrl[b * 3 + 1]));
    float g2 = 1.0f / (1.0f + __expf(-ctrl[b * 3 + 2]));
    float tot = g0 + g1 + g2 + 1e-6f;
    float push = g0 / tot, pop = g1 / tot, stay = g2 / tot;

    if (tid < 64) {
        int s = tid;
        float pu = ptrv[(size_t)b * 64 + ((s + 63) & 63)];   // roll(+1)
        float pd = ptrv[(size_t)b * 64 + ((s + 1) & 63)];    // roll(-1)
        float pc = ptrv[(size_t)b * 64 + s];
        sPtr[s] = push * pu + pop * pd + stay * pc;
        sW[s] = 0.0f;
    }

    const float omp = 1.0f - push;
    for (int i = tid; i < S_ * TWO_D; i += 256) {
        int s = i >> 9, c = i & 511;
        float zc = (c < 256) ? z_real[(size_t)b * 256 + c]
                             : z_imag[(size_t)b * 256 + (c - 256)];
        float v = mem[((size_t)b * S_ + s) * TWO_D + c] * omp + push * zc;
        sM[i] = f2bf(v);
    }
    __syncthreads();

    // ---- Kf = clin_k(mr, mi): 4 row-tiles x 32 col-tiles, WMMA bf16 ----
    for (int tile = wave; tile < 128; tile += 8) {
        int t0 = (tile >> 5) * 16, n0 = (tile & 31) * 16;
        bool isReal = (n0 < 256);
        int nn = isReal ? n0 : (n0 - 256);
        const unsigned short* w1 = wbf + ((size_t)(1 * 3 + (isReal ? 0 : 1)) << 16) + nn * 256;
        const unsigned short* w2 = wbf + ((size_t)(1 * 3 + (isReal ? 2 : 0)) << 16) + nn * 256;
        v8f acc = {};
        for (int k0 = 0; k0 < 256; k0 += 32) {
            FragBF ar = load_a_frag(sM + t0 * TWO_D + k0, TWO_D, lane);
            acc = wmma_bf16(ar, load_b_frag(w1 + k0, 256, lane), acc);
            FragBF ai = load_a_frag(sM + t0 * TWO_D + 256 + k0, TWO_D, lane);
            acc = wmma_bf16(ai, load_b_frag(w2 + k0, 256, lane), acc);
        }
        #pragma unroll
        for (int r = 0; r < 8; ++r)
            sK[(t0 + r + ((lane >> 4) << 3)) * TWO_D + n0 + (lane & 15)] = f2bf(acc[r]);
    }
    __syncthreads();

    // ---- Q tiles, scores, softmax, w accumulation ----
    for (int qt = 0; qt < 4; ++qt) {
        int m0 = qt * 16;
        for (int tc = wave; tc < 32; tc += 8) {
            int n0 = tc * 16;
            bool isReal = (n0 < 256);
            int nn = isReal ? n0 : (n0 - 256);
            const unsigned short* w1 = wbf + ((size_t)(0 * 3 + (isReal ? 0 : 1)) << 16) + nn * 256;
            const unsigned short* w2 = wbf + ((size_t)(0 * 3 + (isReal ? 2 : 0)) << 16) + nn * 256;
            v8f acc = {};
            for (int k0 = 0; k0 < 256; k0 += 32) {
                FragBF ar = load_a_frag(sM + m0 * TWO_D + k0, TWO_D, lane);
                acc = wmma_bf16(ar, load_b_frag(w1 + k0, 256, lane), acc);
                FragBF ai = load_a_frag(sM + m0 * TWO_D + 256 + k0, TWO_D, lane);
                acc = wmma_bf16(ai, load_b_frag(w2 + k0, 256, lane), acc);
            }
            #pragma unroll
            for (int r = 0; r < 8; ++r)
                sQ[(r + ((lane >> 4) << 3)) * TWO_D + n0 + (lane & 15)] = f2bf(acc[r]);
        }
        __syncthreads();

        if (wave < 4) {
            int t0 = wave * 16;
            v8f acc = {};
            for (int k0 = 0; k0 < TWO_D; k0 += 32) {
                FragBF a  = load_a_frag(sQ + k0, TWO_D, lane);
                FragBF bb = load_b_frag(sK + t0 * TWO_D + k0, TWO_D, lane);
                acc = wmma_bf16(a, bb, acc);
            }
            #pragma unroll
            for (int r = 0; r < 8; ++r)
                sS[(r + ((lane >> 4) << 3)) * 64 + t0 + (lane & 15)] = acc[r];
        }
        __syncthreads();

        if (tid < 16) {
            int row = tid;
            float mx = -3.4e38f;
            for (int t = 0; t < 64; ++t) mx = fmaxf(mx, sS[row * 64 + t] * 0.0625f);
            float sum = 0.0f;
            for (int t = 0; t < 64; ++t) {
                float e = __expf(sS[row * 64 + t] * 0.0625f - mx);
                sS[row * 64 + t] = e; sum += e;
            }
            float inv = 1.0f / sum;
            for (int t = 0; t < 64; ++t) sS[row * 64 + t] *= inv;
        }
        __syncthreads();

        if (tid < 64) {
            float acc = 0.0f;
            for (int s = 0; s < 16; ++s) acc += sPtr[m0 + s] * sS[s * 64 + tid];
            sW[tid] += acc;
        }
        __syncthreads();
    }

    for (int c = tid; c < TWO_D; c += 256) {
        float acc = 0.0f;
        for (int t = 0; t < 64; ++t) acc += sW[t] * bf2f(sM[t * TWO_D + c]);
        sU[c] = acc;
    }
    __syncthreads();

    {
        int n = tid;   // 0..255
        float rr = 0.0f, ri = 0.0f;
        const float* wr = Wv_r + (size_t)n * 256;
        const float* wi = Wv_i + (size_t)n * 256;
        for (int k = 0; k < 256; ++k) {
            float ur = sU[k], ui = sU[k + 256];
            float a = wr[k], bw = wi[k];
            rr += ur * a - ui * bw;
            ri += ui * a + ur * bw;
        }
        readf[(size_t)b * TWO_D + n]       = rr;
        readf[(size_t)b * TWO_D + 256 + n] = ri;
        readbf[(size_t)b * TWO_D + n]       = f2bf(rr);
        readbf[(size_t)b * TWO_D + 256 + n] = f2bf(ri);

        float mag = sqrtf(rr * rr + ri * ri);
        sRed[tid] = mag; __syncthreads();
        for (int st = 128; st > 0; st >>= 1) {
            if (tid < st) sRed[tid] += sRed[tid + st];
            __syncthreads();
        }
        float s1 = sRed[0]; __syncthreads();
        sRed[tid] = mag * mag; __syncthreads();
        for (int st = 128; st > 0; st >>= 1) {
            if (tid < st) sRed[tid] += sRed[tid + st];
            __syncthreads();
        }
        float s2 = sRed[0];
        if (tid == 0) {
            rnorm2[b] = s2;
            atomicAdd(&magacc[0], s1);
            atomicAdd(&magacc[1], s2);
        }
    }
}

__global__ __launch_bounds__(1) void k_up_scalar(float* scal) {
    float s1 = scal[0], s2 = scal[1];
    const float M = (float)(B_ * D_);
    float var = (s2 - s1 * s1 / M) / (M - 1.0f);
    float x = var / (1.0f + 1e-6f);
    scal[2] = (x > 20.0f) ? x : log1pf(__expf(x));   // softplus
}

// ---------------------------------------------------------------------------
// logits: 64x64 block tile, 8 waves. Graph GEMM (K=4096) with TDM
// double-buffered LDS staging; content GEMM (K=512) streamed from L2.
// ---------------------------------------------------------------------------
__global__ __launch_bounds__(256) void k_logits(
        const unsigned short* pd_bf, const unsigned short* adjT_bf,
        const unsigned short* read_bf, const unsigned short* cb_bf,
        const float* cnorm, const float* rnorm2, const float* scal,
        float* logits) {
    __shared__ unsigned short sA[2][TDM_TILE_R * LDT];   // prev_dist tiles
    __shared__ unsigned short sB[2][TDM_TILE_R * LDT];   // adjT tiles

    const int n0 = blockIdx.x * 64, b0 = blockIdx.y * 64;
    const int tid = threadIdx.x;
    const int lane = tid & 31;
    const int wave = tid >> 5;
    const int rt  = wave & 3;          // b sub-tile 0..3
    const int ct0 = (wave >> 2) * 2;   // two n sub-tiles

    v8f acc0 = {}, acc1 = {};

    if (wave == 0) {
        tdm_load_2d_bf16(lds_offset(&sA[0][0]), pd_bf   + (size_t)b0 * N_, N_);
        tdm_load_2d_bf16(lds_offset(&sB[0][0]), adjT_bf + (size_t)n0 * N_, N_);
    }
    const int STEPS = N_ / TDM_TILE_K;   // 128
    for (int k = 0; k < STEPS; ++k) {
        if (wave == 0) {
            if (k + 1 < STEPS) {
                int kk = (k + 1) * TDM_TILE_K;
                tdm_load_2d_bf16(lds_offset(&sA[(k + 1) & 1][0]), pd_bf   + (size_t)b0 * N_ + kk, N_);
                tdm_load_2d_bf16(lds_offset(&sB[(k + 1) & 1][0]), adjT_bf + (size_t)n0 * N_ + kk, N_);
                __builtin_amdgcn_s_wait_tensorcnt(2);    // step-k pair complete
            } else {
                __builtin_amdgcn_s_wait_tensorcnt(0);
            }
        }
        __syncthreads();                                 // publish buf[k&1]
        const unsigned short* a = &sA[k & 1][0] + rt * 16 * LDT;
        const unsigned short* bp = &sB[k & 1][0];
        FragBF af = load_a_frag(a, LDT, lane);
        acc0 = wmma_bf16(af, load_b_frag(bp + (ct0    ) * 16 * LDT, LDT, lane), acc0);
        acc1 = wmma_bf16(af, load_b_frag(bp + (ct0 + 1) * 16 * LDT, LDT, lane), acc1);
        __syncthreads();                                 // done reading buf[k&1]
    }

    // content GEMM: read @ codebook^T, K=512 (L2 resident)
    v8f dc0 = {}, dc1 = {};
    for (int k0 = 0; k0 < TWO_D; k0 += 32) {
        FragBF a = load_a_frag(read_bf + (size_t)(b0 + rt * 16) * TWO_D + k0, TWO_D, lane);
        dc0 = wmma_bf16(a, load_b_frag(cb_bf + (size_t)(n0 + ct0 * 16) * TWO_D + k0, TWO_D, lane), dc0);
        dc1 = wmma_bf16(a, load_b_frag(cb_bf + (size_t)(n0 + (ct0 + 1) * 16) * TWO_D + k0, TWO_D, lane), dc1);
    }

    float up = scal[2];
    #pragma unroll
    for (int t = 0; t < 2; ++t) {
        v8f g  = t ? acc1 : acc0;
        v8f dc = t ? dc1 : dc0;
        int col = n0 + (ct0 + t) * 16 + (lane & 15);
        float cn = cnorm[col];
        #pragma unroll
        for (int r = 0; r < 8; ++r) {
            int row = b0 + rt * 16 + r + ((lane >> 4) << 3);
            float dcon = (rnorm2[row] + cn - 2.0f * dc[r]) * (1.0f / 512.0f);
            float sg = 1.0f / (1.0f + __expf(-g[r]));
            logits[(size_t)row * N_ + col] = -(dcon - 0.1f * up * sg);   // TEMP=1
        }
    }
}

__global__ __launch_bounds__(256) void k_softmax(const float* logits,
                                                 unsigned short* probs_bf) {
    __shared__ float red[256];
    int b = blockIdx.x, tid = threadIdx.x;
    const float* row = logits + (size_t)b * N_;
    float mx = -3.4e38f;
    for (int i = tid; i < N_; i += 256) mx = fmaxf(mx, row[i]);
    red[tid] = mx; __syncthreads();
    for (int st = 128; st > 0; st >>= 1) {
        if (tid < st) red[tid] = fmaxf(red[tid], red[tid + st]);
        __syncthreads();
    }
    mx = red[0]; __syncthreads();
    float sum = 0.0f;
    for (int i = tid; i < N_; i += 256) sum += __expf(row[i] - mx);
    red[tid] = sum; __syncthreads();
    for (int st = 128; st > 0; st >>= 1) {
        if (tid < st) red[tid] += red[tid + st];
        __syncthreads();
    }
    float inv = 1.0f / red[0];
    for (int i = tid; i < N_; i += 256)
        probs_bf[(size_t)b * N_ + i] = f2bf(__expf(row[i] - mx) * inv);
}

// ---------------------------------------------------------------------------
// z_q = probs @ codebook  (B x N) @ (N x 512): 64x64 block tile, 8 waves,
// TDM double-buffered staging, f32 out straight to d_out.
// ---------------------------------------------------------------------------
__global__ __launch_bounds__(256) void k_zq(const unsigned short* probs_bf,
                                            const unsigned short* cbT_bf,
                                            float* out) {
    __shared__ unsigned short sA[2][TDM_TILE_R * LDT];   // probs tiles
    __shared__ unsigned short sB[2][TDM_TILE_R * LDT];   // codebook^T tiles

    const int c0 = blockIdx.x * 64, b0 = blockIdx.y * 64;
    const int tid = threadIdx.x;
    const int lane = tid & 31;
    const int wave = tid >> 5;
    const int rt  = wave & 3;
    const int ct0 = (wave >> 2) * 2;

    v8f acc0 = {}, acc1 = {};

    if (wave == 0) {
        tdm_load_2d_bf16(lds_offset(&sA[0][0]), probs_bf + (size_t)b0 * N_, N_);
        tdm_load_2d_bf16(lds_offset(&sB[0][0]), cbT_bf   + (size_t)c0 * N_, N_);
    }
    const int STEPS = N_ / TDM_TILE_K;
    for (int k = 0; k < STEPS; ++k) {
        if (wave == 0) {
            if (k + 1 < STEPS) {
                int kk = (k + 1) * TDM_TILE_K;
                tdm_load_2d_bf16(lds_offset(&sA[(k + 1) & 1][0]), probs_bf + (size_t)b0 * N_ + kk, N_);
                tdm_load_2d_bf16(lds_offset(&sB[(k + 1) & 1][0]), cbT_bf   + (size_t)c0 * N_ + kk, N_);
                __builtin_amdgcn_s_wait_tensorcnt(2);
            } else {
                __builtin_amdgcn_s_wait_tensorcnt(0);
            }
        }
        __syncthreads();
        const unsigned short* a = &sA[k & 1][0] + rt * 16 * LDT;
        const unsigned short* bp = &sB[k & 1][0];
        FragBF af = load_a_frag(a, LDT, lane);
        acc0 = wmma_bf16(af, load_b_frag(bp + (ct0    ) * 16 * LDT, LDT, lane), acc0);
        acc1 = wmma_bf16(af, load_b_frag(bp + (ct0 + 1) * 16 * LDT, LDT, lane), acc1);
        __syncthreads();
    }

    #pragma unroll
    for (int t = 0; t < 2; ++t) {
        v8f acc = t ? acc1 : acc0;
        int col = c0 + (ct0 + t) * 16 + (lane & 15);
        #pragma unroll
        for (int r = 0; r < 8; ++r) {
            int row = b0 + rt * 16 + r + ((lane >> 4) << 3);
            out[(size_t)row * TWO_D + col] = acc[r];
        }
    }
}

// ---------------------------------------------------------------------------
extern "C" void kernel_launch(void* const* d_in, const int* in_sizes, int n_in,
                              void* d_out, int out_size, void* d_ws, size_t ws_size,
                              hipStream_t stream) {
    const float* z_real    = (const float*)d_in[0];
    const float* z_imag    = (const float*)d_in[1];
    const float* mem       = (const float*)d_in[2];
    const float* ptrv      = (const float*)d_in[3];
    const float* ctrl      = (const float*)d_in[4];
    const float* prev_dist = (const float*)d_in[5];
    const float* Wq_r      = (const float*)d_in[6];
    const float* Wq_i      = (const float*)d_in[7];
    const float* Wk_r      = (const float*)d_in[8];
    const float* Wk_i      = (const float*)d_in[9];
    const float* Wv_r      = (const float*)d_in[10];
    const float* Wv_i      = (const float*)d_in[11];
    const float* codebook  = (const float*)d_in[12];
    const float* adjacency = (const float*)d_in[13];

    char* ws = (char*)d_ws;
    unsigned short* wbf      = (unsigned short*)(ws + OFF_WBF);
    unsigned short* adjT_bf  = (unsigned short*)(ws + OFF_ADJT);
    unsigned short* pd_bf    = (unsigned short*)(ws + OFF_PD);
    unsigned short* probs_bf = (unsigned short*)(ws + OFF_PROBS);
    unsigned short* cb_bf    = (unsigned short*)(ws + OFF_CB);
    unsigned short* cbT_bf   = (unsigned short*)(ws + OFF_CBT);
    unsigned short* read_bf  = (unsigned short*)(ws + OFF_RBF);
    float*          readf    = (float*)(ws + OFF_RF);
    float*          cnorm    = (float*)(ws + OFF_CNORM);
    float*          rnorm2   = (float*)(ws + OFF_RNORM);
    float*          scal     = (float*)(ws + OFF_SCAL);
    float*          logits   = (float*)(ws + OFF_LOGITS);

    k_zero_scalars<<<1, 1, 0, stream>>>(scal);
    k_prep_weights<<<(2 * 3 * 65536) / 256, 256, 0, stream>>>(Wq_r, Wq_i, Wk_r, Wk_i, wbf);
    k_prep_pd<<<(B_ * N_) / 256, 256, 0, stream>>>(prev_dist, pd_bf);
    k_prep_adjT<<<dim3(N_ / 64, N_ / 64), 256, 0, stream>>>(adjacency, adjT_bf);
    k_prep_cb<<<N_, 256, 0, stream>>>(codebook, cb_bf, cbT_bf, cnorm);

    k_batch_attn<<<B_, 256, 0, stream>>>(z_real, z_imag, mem, ptrv, ctrl,
                                         wbf, Wv_r, Wv_i,
                                         readf, read_bf, rnorm2, scal);
    k_up_scalar<<<1, 1, 0, stream>>>(scal);

    k_logits<<<dim3(N_ / 64, B_ / 64), 256, 0, stream>>>(pd_bf, adjT_bf, read_bf,
                                                         cb_bf, cnorm, rnorm2, scal,
                                                         logits);
    k_softmax<<<B_, 256, 0, stream>>>(logits, probs_bf);
    k_zq<<<dim3(TWO_D / 64, B_ / 64), 256, 0, stream>>>(probs_bf, cbT_bf, (float*)d_out);
}
